// AttentionBlock_13305808683222
// MI455X (gfx1250) — matrix operand order repaired and checked by reading, not compiled, not run on hardware
//
#include <hip/hip_runtime.h>
#include <cstdint>
#include <cstddef>

typedef _Float16 v16h __attribute__((ext_vector_type(16)));
typedef _Float16 v8h  __attribute__((ext_vector_type(8)));
typedef float    v8f  __attribute__((ext_vector_type(8)));

#define DIMK 1024
#define SEQ  2048
#define NB   4

#define BM 128
#define BN 128
#define BK 32
#define LDP 40   // padded LDS row pitch in halves (80B -> conflict-free b128 reads)

// ---------------- split fp32 -> (hi,lo) f16 planes ----------------
__global__ __launch_bounds__(256) void split_kernel(const float* __restrict__ src,
                                                    _Float16* __restrict__ hi,
                                                    _Float16* __restrict__ lo,
                                                    int n) {
  int i = blockIdx.x * 256 + threadIdx.x;
  if (i < n) {
    float v = src[i];
    _Float16 h = (_Float16)v;
    hi[i] = h;
    lo[i] = (_Float16)(v - (float)h);
  }
}

// ---------------- fragment loader: two 16B LDS reads -> v16h ----------------
static __device__ inline v16h ld16(const _Float16* p, int o0, int o1) {
  v8h a = *(const v8h*)(p + o0);
  v8h b = *(const v8h*)(p + o1);
  return __builtin_shufflevector(a, b, 0,1,2,3,4,5,6,7,8,9,10,11,12,13,14,15);
}

// ---------------- generic split-f16 GEMM, register-prefetch pipelined ----------------
// C[m][n] = sum_k (Ah+Al)[m][k] * (Bh+Bl)[n][k]  (+bias[n]) , scaled by outScale
// mode 0: outF[m*ldo+n] = v (+resid[m*ldo+n])
// mode 1: split-store to outH/outL[m*ldo+n]
// mode 2: split-store transposed to outH/outL[n*ldt+m]
__global__ __launch_bounds__(256) void gemm_sp(
    const _Float16* __restrict__ Ah, const _Float16* __restrict__ Al, int lda,
    const _Float16* __restrict__ Bh, const _Float16* __restrict__ Bl, int ldb,
    const float* __restrict__ bias,
    const float* __restrict__ resid,
    float* __restrict__ outF,
    _Float16* __restrict__ outH, _Float16* __restrict__ outL,
    int ldo, int ldt,
    int K, float outScale, int mode)
{
  __shared__ _Float16 As[2][BM * LDP];
  __shared__ _Float16 Bs[2][BN * LDP];

  const int t    = threadIdx.x;
  const int lane = t & 31;
  const int lr   = lane & 15;   // N (or M) position within fragment
  const int lh   = lane >> 4;   // half-wave select
  const int w    = t >> 5;      // wave id 0..7
  const int wn   = w & 3;       // 4 waves along N (32 cols each)
  const int wm   = w >> 2;      // 2 waves along M (64 rows each)

  const int m0 = blockIdx.y * BM;
  const int n0 = blockIdx.x * BN;

  // staging geometry for this thread (two 16B chunks per plane per matrix)
  const int c0row = t >> 2, c0off = (t & 3) * 8;                  // chunk t
  const int c1row = (t + 256) >> 2, c1off = ((t + 256) & 3) * 8;  // chunk t+256

  // 32-bit per-thread offsets; uniform 64-bit plane bases stay in SGPRs
  const unsigned offA0 = (unsigned)(m0 + c0row) * (unsigned)lda + c0off;
  const unsigned offA1 = (unsigned)(m0 + c1row) * (unsigned)lda + c1off;
  const unsigned offB0 = (unsigned)(n0 + c0row) * (unsigned)ldb + c0off;
  const unsigned offB1 = (unsigned)(n0 + c1row) * (unsigned)ldb + c1off;

  v8f acc[4][2];
  #pragma unroll
  for (int mt = 0; mt < 4; ++mt) {
    #pragma unroll
    for (int nt = 0; nt < 2; ++nt) {
      float b0 = bias ? bias[n0 + wn*32 + nt*16 + lr] : 0.0f;
      #pragma unroll
      for (int e = 0; e < 8; ++e) acc[mt][nt][e] = b0;
    }
  }

  uint4 pa[2][2], pb[2][2];   // [plane][chunk]
  // prefetch K-chunk 0
  pa[0][0] = *(const uint4*)(Ah + offA0);
  pa[0][1] = *(const uint4*)(Ah + offA1);
  pa[1][0] = *(const uint4*)(Al + offA0);
  pa[1][1] = *(const uint4*)(Al + offA1);
  pb[0][0] = *(const uint4*)(Bh + offB0);
  pb[0][1] = *(const uint4*)(Bh + offB1);
  pb[1][0] = *(const uint4*)(Bl + offB0);
  pb[1][1] = *(const uint4*)(Bl + offB1);

  for (int kc = 0; kc < K; kc += BK) {
    __syncthreads();
    // deposit prefetched chunk into LDS
    #pragma unroll
    for (int p = 0; p < 2; ++p) {
      *(uint4*)(&As[p][c0row * LDP + c0off]) = pa[p][0];
      *(uint4*)(&As[p][c1row * LDP + c1off]) = pa[p][1];
      *(uint4*)(&Bs[p][c0row * LDP + c0off]) = pb[p][0];
      *(uint4*)(&Bs[p][c1row * LDP + c1off]) = pb[p][1];
    }
    __syncthreads();

    // issue next chunk's global loads while computing this one
    if (kc + BK < K) {
      const unsigned ko = (unsigned)(kc + BK);
      pa[0][0] = *(const uint4*)(Ah + offA0 + ko);
      pa[0][1] = *(const uint4*)(Ah + offA1 + ko);
      pa[1][0] = *(const uint4*)(Al + offA0 + ko);
      pa[1][1] = *(const uint4*)(Al + offA1 + ko);
      pb[0][0] = *(const uint4*)(Bh + offB0 + ko);
      pb[0][1] = *(const uint4*)(Bh + offB1 + ko);
      pb[1][0] = *(const uint4*)(Bl + offB0 + ko);
      pb[1][1] = *(const uint4*)(Bl + offB1 + ko);
    }

    // B fragments: lane holds col n=lr, K runs {8lh.., 16+8lh..}
    v16h bf[2][2];
    #pragma unroll
    for (int nt = 0; nt < 2; ++nt) {
      const int nr = (wn*32 + nt*16 + lr) * LDP;
      #pragma unroll
      for (int p = 0; p < 2; ++p)
        bf[nt][p] = ld16(&Bs[p][nr], 8*lh, 16 + 8*lh);
    }

    #pragma unroll
    for (int mt = 0; mt < 4; ++mt) {
      // A fragments: lane holds row m=lr, K runs {8lh.., 16+8lh..}
      const int mr = (wm*64 + mt*16 + lr) * LDP;
      v16h ah = ld16(&As[0][mr], 8*lh, 16 + 8*lh);
      v16h al = ld16(&As[1][mr], 8*lh, 16 + 8*lh);
      #pragma unroll
      for (int nt = 0; nt < 2; ++nt) {
        acc[mt][nt] = __builtin_amdgcn_wmma_f32_16x16x32_f16(
            false, ah, false, bf[nt][0], (short)0, acc[mt][nt], false, false);
        acc[mt][nt] = __builtin_amdgcn_wmma_f32_16x16x32_f16(
            false, ah, false, bf[nt][1], (short)0, acc[mt][nt], false, false);
        acc[mt][nt] = __builtin_amdgcn_wmma_f32_16x16x32_f16(
            false, al, false, bf[nt][0], (short)0, acc[mt][nt], false, false);
      }
    }
  }

  // epilogue: C/D layout -> row = e + 8*lh, col = lr
  #pragma unroll
  for (int mt = 0; mt < 4; ++mt) {
    #pragma unroll
    for (int nt = 0; nt < 2; ++nt) {
      const int gn = n0 + wn*32 + nt*16 + lr;
      const int rb = m0 + wm*64 + mt*16 + 8*lh;
      #pragma unroll
      for (int e = 0; e < 8; ++e) {
        float v = acc[mt][nt][e] * outScale;
        const int gm = rb + e;
        if (mode == 0) {
          float r = resid ? resid[(size_t)gm * ldo + gn] : 0.0f;
          outF[(size_t)gm * ldo + gn] = v + r;
        } else if (mode == 1) {
          _Float16 h = (_Float16)v;
          outH[(size_t)gm * ldo + gn] = h;
          outL[(size_t)gm * ldo + gn] = (_Float16)(v - (float)h);
        } else {
          _Float16 h = (_Float16)v;
          outH[(size_t)gn * ldt + gm] = h;
          outL[(size_t)gn * ldt + gm] = (_Float16)(v - (float)h);
        }
      }
    }
  }
}

// ---------------- row softmax, overwriting the f32 row with P hi/lo f16 planes ----------------
__global__ __launch_bounds__(256) void softmax_rows(float* __restrict__ S) {
  __shared__ float buf[SEQ];
  __shared__ float red[256];
  const int t = threadIdx.x;
  float* row = S + (size_t)blockIdx.x * SEQ;

  float m = -3.4e38f;
  #pragma unroll
  for (int i = 0; i < SEQ / 256; ++i) {
    float v = row[t + i * 256];
    buf[t + i * 256] = v;
    m = fmaxf(m, v);
  }
  red[t] = m;
  __syncthreads();
  for (int s = 128; s > 0; s >>= 1) {
    if (t < s) red[t] = fmaxf(red[t], red[t + s]);
    __syncthreads();
  }
  m = red[0];
  __syncthreads();

  float sum = 0.0f;
  #pragma unroll
  for (int i = 0; i < SEQ / 256; ++i) sum += expf(buf[t + i * 256] - m);
  red[t] = sum;
  __syncthreads();
  for (int s = 128; s > 0; s >>= 1) {
    if (t < s) red[t] += red[t + s];
    __syncthreads();
  }
  const float inv = 1.0f / red[0];

  // overwrite row memory: first SEQ halves = hi plane, next SEQ halves = lo plane
  _Float16* ph = (_Float16*)row;
  #pragma unroll
  for (int i = 0; i < SEQ / 256; ++i) {
    int j = t + i * 256;
    float p = expf(buf[j] - m) * inv;
    _Float16 h = (_Float16)p;
    ph[j] = h;
    ph[SEQ + j] = (_Float16)(p - (float)h);
  }
}

// ---------------- host launch ----------------
extern "C" void kernel_launch(void* const* d_in, const int* in_sizes, int n_in,
                              void* d_out, int out_size, void* d_ws, size_t ws_size,
                              hipStream_t stream) {
  (void)in_sizes; (void)n_in; (void)out_size; (void)ws_size;

  const float* x   = (const float*)d_in[0];
  const float* Wq  = (const float*)d_in[1];
  const float* bq  = (const float*)d_in[2];
  const float* Wk  = (const float*)d_in[3];
  const float* bk  = (const float*)d_in[4];
  const float* Wv  = (const float*)d_in[5];
  const float* bv  = (const float*)d_in[6];
  const float* Wfc = (const float*)d_in[7];
  const float* bfc = (const float*)d_in[8];
  float* out = (float*)d_out;

  const size_t R       = (size_t)NB * SEQ;     // 8192 rows
  const size_t PLANE_X = R * DIMK;             // 8,388,608 elems
  const size_t PLANE_W = (size_t)DIMK * DIMK;  // 1,048,576 elems

  char*  ws  = (char*)d_ws;
  size_t off = 0;
  auto alloc = [&](size_t bytes) -> char* {
    char* p = ws + off;
    off += (bytes + 255) & ~(size_t)255;
    return p;
  };

  _Float16* xh  = (_Float16*)alloc(PLANE_X * 2);
  _Float16* xl  = (_Float16*)alloc(PLANE_X * 2);
  _Float16* wqh = (_Float16*)alloc(PLANE_W * 2);
  _Float16* wql = (_Float16*)alloc(PLANE_W * 2);
  _Float16* wkh = (_Float16*)alloc(PLANE_W * 2);
  _Float16* wkl = (_Float16*)alloc(PLANE_W * 2);
  _Float16* wvh = (_Float16*)alloc(PLANE_W * 2);
  _Float16* wvl = (_Float16*)alloc(PLANE_W * 2);
  _Float16* wfh = (_Float16*)alloc(PLANE_W * 2);
  _Float16* wfl = (_Float16*)alloc(PLANE_W * 2);
  _Float16* qh  = (_Float16*)alloc(PLANE_X * 2);
  _Float16* ql  = (_Float16*)alloc(PLANE_X * 2);
  _Float16* kh  = (_Float16*)alloc(PLANE_X * 2);
  _Float16* kl  = (_Float16*)alloc(PLANE_X * 2);
  _Float16* vth = (_Float16*)alloc(PLANE_X * 2);   // V transposed: [d][r], ldt = R
  _Float16* vtl = (_Float16*)alloc(PLANE_X * 2);
  float*    Sbuf = (float*)alloc((size_t)SEQ * SEQ * 4);  // per-batch, reused

  // O planes alias x planes (x planes are dead after the QKV projections)
  _Float16* oh = xh;
  _Float16* ol = xl;

  dim3 blk(256);

  // 1) split inputs to f16 hi/lo planes
  split_kernel<<<dim3((unsigned)((PLANE_X + 255) / 256)), blk, 0, stream>>>(x,   xh,  xl,  (int)PLANE_X);
  split_kernel<<<dim3((unsigned)((PLANE_W + 255) / 256)), blk, 0, stream>>>(Wq,  wqh, wql, (int)PLANE_W);
  split_kernel<<<dim3((unsigned)((PLANE_W + 255) / 256)), blk, 0, stream>>>(Wk,  wkh, wkl, (int)PLANE_W);
  split_kernel<<<dim3((unsigned)((PLANE_W + 255) / 256)), blk, 0, stream>>>(Wv,  wvh, wvl, (int)PLANE_W);
  split_kernel<<<dim3((unsigned)((PLANE_W + 255) / 256)), blk, 0, stream>>>(Wfc, wfh, wfl, (int)PLANE_W);

  // 2) projections: M=8192, N=1024, K=1024
  dim3 gproj(DIMK / BN, (unsigned)(R / BM), 1);  // (8, 64)
  // Q (scale 1/sqrt(1024)=1/32 folded in), K, V (transposed store)
  gemm_sp<<<gproj, blk, 0, stream>>>(xh, xl, DIMK, wqh, wql, DIMK, bq, nullptr,
                                     nullptr, qh, ql, DIMK, 0, DIMK, 1.0f / 32.0f, 1);
  gemm_sp<<<gproj, blk, 0, stream>>>(xh, xl, DIMK, wkh, wkl, DIMK, bk, nullptr,
                                     nullptr, kh, kl, DIMK, 0, DIMK, 1.0f, 1);
  gemm_sp<<<gproj, blk, 0, stream>>>(xh, xl, DIMK, wvh, wvl, DIMK, bv, nullptr,
                                     nullptr, vth, vtl, 0, (int)R, DIMK, 1.0f, 2);

  // 3) attention, one batch at a time (S buffer reused)
  for (int b = 0; b < NB; ++b) {
    const size_t ro = (size_t)b * SEQ * DIMK;  // row offset into [r][d] planes
    // S = Q_b @ K_b^T : M=N=2048, K=1024
    dim3 gs(SEQ / BN, SEQ / BM, 1);  // (16,16)
    gemm_sp<<<gs, blk, 0, stream>>>(qh + ro, ql + ro, DIMK, kh + ro, kl + ro, DIMK,
                                    nullptr, nullptr, Sbuf, nullptr, nullptr,
                                    SEQ, 0, DIMK, 1.0f, 0);
    // softmax per row; rows rewritten in place as P hi/lo f16 planes
    softmax_rows<<<dim3(SEQ), blk, 0, stream>>>(Sbuf);
    // O = P @ V : M=2048, N=1024, K=2048
    const _Float16* pA  = (const _Float16*)Sbuf;       // row pitch 2*SEQ halves
    dim3 gpv(DIMK / BN, SEQ / BM, 1);  // (8,16)
    gemm_sp<<<gpv, blk, 0, stream>>>(pA, pA + SEQ, 2 * SEQ,
                                     vth + (size_t)b * SEQ, vtl + (size_t)b * SEQ, (int)R,
                                     nullptr, nullptr,
                                     nullptr, oh + ro, ol + ro, DIMK, 0, SEQ, 1.0f, 1);
  }

  // 4) FC + bias + residual: out = O @ Wfc^T + bfc + x
  gemm_sp<<<gproj, blk, 0, stream>>>(oh, ol, DIMK, wfh, wfl, DIMK, bfc, x,
                                     out, nullptr, nullptr, DIMK, 0, DIMK, 1.0f, 0);
}
